// SparseLinear_30777735643891
// MI455X (gfx1250) — matrix operand order repaired and checked by reading, not compiled, not run on hardware
//
#include <hip/hip_runtime.h>
#include <cstdint>

// ---------------------------------------------------------------------------
// SparseLinear for MI455X (gfx1250):
//   out = x @ W^T + bias, W scattered from COO (5% dense) -> dense bf16.
// Phase 1: memset 32MB ws, scatter bf16 weights directly into the WMMA
//          B-fragment-linear swizzled layout (so GEMM B-tiles are contiguous
//          16KB blocks, streamed with global_load_async_to_lds_b128).
// Phase 2: bf16 WMMA GEMM, f32 accumulate, K-step 32 (v_wmma_f32_16x16x32_bf16),
//          double-buffered LDS, async B staging, HW packed-cvt A staging.
// ---------------------------------------------------------------------------

#define BATCH   16384
#define CH      4096            // IN_CH == OUT_CH
#define KSTEPS  (CH / 32)       // 128 K-blocks of 32
#define NPANELS (CH / 256)      // 16 N-panels of 256
#define MTILES  (BATCH / 128)   // 128 M-tiles of 128

typedef __attribute__((ext_vector_type(16))) __bf16 v16bf;
typedef __attribute__((ext_vector_type(2)))  __bf16 v2bf;
typedef __attribute__((ext_vector_type(8)))  float  v8f;
typedef __attribute__((ext_vector_type(4)))  int    v4i;

typedef __attribute__((address_space(1))) v4i* gas_v4i_p;   // global
typedef __attribute__((address_space(3))) v4i* las_v4i_p;   // LDS

// f32 -> bf16 (RNE) through the hardware converter, not bit twiddling.
__device__ __forceinline__ unsigned pack2(float lo, float hi) {
#if defined(__has_builtin) && __has_builtin(__builtin_amdgcn_cvt_pk_bf16_f32)
    union { v2bf v; unsigned u; } c;
    c.v = __builtin_amdgcn_cvt_pk_bf16_f32(lo, hi);   // v_cvt_pk_bf16_f32
    return c.u;
#else
    union { __bf16 b[2]; unsigned u; } c;
    c.b[0] = (__bf16)lo; c.b[1] = (__bf16)hi;         // fptrunc -> HW cvt
    return c.u;
#endif
}
__device__ __forceinline__ unsigned short f32_to_bf16(float f) {
    union { __bf16 b; unsigned short u; } c;
    c.b = (__bf16)f;
    return c.u;
}
// XOR-swizzle of 16B-block index inside a 1KB fragment tile: keeps each lane's
// 32B fragment contiguous/aligned but spreads lanes L, L+8, L+16, L+24 across
// distinct bank groups (plain lane*32 layout is a 4-way conflict).
__device__ __forceinline__ unsigned swz(unsigned b) { return b ^ ((b >> 4) << 1); }

__device__ __forceinline__ void async_b128(const void* g, void* l) {
#if defined(__has_builtin) && __has_builtin(__builtin_amdgcn_global_load_async_to_lds_b128)
    __builtin_amdgcn_global_load_async_to_lds_b128(
        (gas_v4i_p)(uintptr_t)g,
        (las_v4i_p)(unsigned)(uintptr_t)l,
        0, 0);
#else
    unsigned lds = (unsigned)(uintptr_t)l;              // low 32 bits = LDS byte addr
    unsigned long long ga = (unsigned long long)(uintptr_t)g;
    asm volatile("global_load_async_to_lds_b128 %0, %1, off" :: "v"(lds), "v"(ga) : "memory");
#endif
}
__device__ __forceinline__ void wait_async0() {
#if defined(__has_builtin) && __has_builtin(__builtin_amdgcn_s_wait_asynccnt)
    __builtin_amdgcn_s_wait_asynccnt(0);
#else
    asm volatile("s_wait_asynccnt 0x0" ::: "memory");
#endif
}

// --------------------------- Phase 1: scatter ------------------------------
// W_swz layout (bf16), fragment-linear for V_WMMA B operand (K=32 x N=16):
//   tileIdx = ((o>>8)*KSTEPS + (c>>5))*16 + ((o>>4)&15)     [N-panel, K-blk, N-col]
//   lane    = (o&15) + 16*((c>>4)&1)                        [B: lanes16-31 hold K16-31]
//   elem    = tileIdx*512 + lane*16 + (c&15)
__global__ __launch_bounds__(256) void sl_scatter(
    const float* __restrict__ w, const int* __restrict__ rows,
    const int* __restrict__ cols, unsigned short* __restrict__ Wswz, int nnz) {
    int i = blockIdx.x * 256 + threadIdx.x;
    if (i >= nnz) return;
    unsigned r = ((unsigned)rows[i]) & (CH - 1);
    unsigned c = ((unsigned)cols[i]) & (CH - 1);
    unsigned tileIdx = ((r >> 8) * KSTEPS + (c >> 5)) * 16u + ((r >> 4) & 15u);
    unsigned lane = (r & 15u) + (((c >> 4) & 1u) << 4);
    Wswz[tileIdx * 512u + lane * 16u + (c & 15u)] = f32_to_bf16(w[i]);
}

// ----------------------------- Phase 2: GEMM -------------------------------
__global__ __launch_bounds__(256) void sl_gemm(
    const float* __restrict__ x, const unsigned short* __restrict__ Wswz,
    const float* __restrict__ bias, float* __restrict__ out) {
    // LDS: A0[8K] @0, B0[16K] @8192, A1[8K] @24576, B1[16K] @32768
    __shared__ __align__(128) unsigned char smem[49152];

    const unsigned tid  = threadIdx.x;
    const unsigned lane = tid & 31u, wid = tid >> 5;
    const unsigned wm = wid >> 2, wn = wid & 3u;       // 2 x 4 wave grid
    const unsigned P  = blockIdx.x & (NPANELS - 1);    // N-panel varies fastest:
    const unsigned mt = blockIdx.x >> 4;               // x M-band stays L2-resident
    const unsigned m0 = mt * 128u;

    // Per-thread A staging: two 8-float chunks per K-step. chunk ci = m*4 + k8.
    const unsigned ci0 = tid, ci1 = tid + 256u;
    const unsigned mA0 = ci0 >> 2, k80 = ci0 & 3u;
    const unsigned mA1 = ci1 >> 2, k81 = ci1 & 3u;
    const float* xp0 = x + (size_t)(m0 + mA0) * CH + k80 * 8u;
    const float* xp1 = x + (size_t)(m0 + mA1) * CH + k81 * 8u;
    // A fragment layout (16x32 bf16): lane = m%16 + 16*(k>>4),
    // byte = (k&15)*2  -> chunk (m,k8) is one 16B block:
    const unsigned dA0 = (mA0 >> 4) * 1024u + swz(2u * ((mA0 & 15u) + ((k80 >> 1) << 4)) + (k80 & 1u)) * 16u;
    const unsigned dA1 = (mA1 >> 4) * 1024u + swz(2u * ((mA1 & 15u) + ((k81 >> 1) << 4)) + (k81 & 1u)) * 16u;
    const unsigned fragOff = (2u * lane ^ ((lane >> 3) << 1)) * 16u;  // lane's 32B frag

    v8f acc[16] = {};

    const unsigned char* Wbase = (const unsigned char*)Wswz + (size_t)P * KSTEPS * 16384u;

#define ISSUE_B(kb, Bbase)                                                         \
    {                                                                              \
        const unsigned char* gB = Wbase + (size_t)(kb) * 16384u;                   \
        _Pragma("unroll") for (unsigned j = 0; j < 4; ++j) {                       \
            unsigned ci = tid + j * 256u;                                          \
            async_b128(gB + ci * 16u,                                              \
                       smem + (Bbase) + (ci >> 6) * 1024u + swz(ci & 63u) * 16u);  \
        }                                                                          \
    }
#define STORE_A(f0a, f0b, f1a, f1b, Abase)                                         \
    {                                                                              \
        uint4 d0 = { pack2(f0a.x, f0a.y), pack2(f0a.z, f0a.w),                     \
                     pack2(f0b.x, f0b.y), pack2(f0b.z, f0b.w) };                   \
        uint4 d1 = { pack2(f1a.x, f1a.y), pack2(f1a.z, f1a.w),                     \
                     pack2(f1b.x, f1b.y), pack2(f1b.z, f1b.w) };                   \
        *reinterpret_cast<uint4*>(smem + (Abase) + dA0) = d0;                      \
        *reinterpret_cast<uint4*>(smem + (Abase) + dA1) = d1;                      \
    }

    // Prologue: stage K-block 0 into buffer set 0.
    {
        const float4* s0 = reinterpret_cast<const float4*>(xp0);
        const float4* s1 = reinterpret_cast<const float4*>(xp1);
        float4 a0 = s0[0], b0 = s0[1], a1 = s1[0], b1 = s1[1];
        ISSUE_B(0, 8192u);
        STORE_A(a0, b0, a1, b1, 0u);
        wait_async0();
        __syncthreads();
    }

    for (int kb = 0; kb < KSTEPS; ++kb) {
        const unsigned cur = (unsigned)kb & 1u;
        const unsigned Ac = cur ? 24576u : 0u,    Bc = cur ? 32768u : 8192u;
        const unsigned An = cur ? 0u     : 24576u, Bn = cur ? 8192u  : 32768u;

        float4 a0, b0, a1, b1;
        const bool more = (kb + 1) < KSTEPS;
        if (more) {
            ISSUE_B(kb + 1, Bn);  // async engine fills next B tile
            const float4* s0 = reinterpret_cast<const float4*>(xp0 + (size_t)(kb + 1) * 32u);
            const float4* s1 = reinterpret_cast<const float4*>(xp1 + (size_t)(kb + 1) * 32u);
            a0 = s0[0]; b0 = s0[1]; a1 = s1[0]; b1 = s1[1];
        }

        // Fragment loads + 16 WMMAs on current buffers (64x64 wave tile).
        v16bf af[4], bf[4];
#pragma unroll
        for (unsigned i = 0; i < 4; ++i)
            af[i] = *reinterpret_cast<const v16bf*>(smem + Ac + (wm * 4u + i) * 1024u + fragOff);
#pragma unroll
        for (unsigned j = 0; j < 4; ++j)
            bf[j] = *reinterpret_cast<const v16bf*>(smem + Bc + (wn * 4u + j) * 1024u + fragOff);
#pragma unroll
        for (unsigned i = 0; i < 4; ++i)
#pragma unroll
            for (unsigned j = 0; j < 4; ++j)
                acc[i * 4u + j] = __builtin_amdgcn_wmma_f32_16x16x32_bf16(
                    false, af[i], false, bf[j], (short)0, acc[i * 4u + j], false, false);

        if (more) STORE_A(a0, b0, a1, b1, An);
        wait_async0();
        __syncthreads();
    }

    // Epilogue: C/D layout -> VGPR r: lanes0-15 (M=r,N=lane), lanes16-31 (M=r+8).
    const unsigned nlo = lane & 15u, mhi = (lane >> 4) * 8u;
#pragma unroll
    for (unsigned i = 0; i < 4; ++i) {
        const unsigned mb = m0 + wm * 64u + i * 16u + mhi;
#pragma unroll
        for (unsigned j = 0; j < 4; ++j) {
            const unsigned n = P * 256u + (wn * 4u + j) * 16u + nlo;
            const float bv = bias[n];
            float* o = out + (size_t)mb * CH + n;
#pragma unroll
            for (unsigned r = 0; r < 8; ++r)
                o[(size_t)r * CH] = acc[i * 4u + j][r] + bv;
        }
    }
#undef ISSUE_B
#undef STORE_A
}

// ------------------------------- Launcher ----------------------------------
extern "C" void kernel_launch(void* const* d_in, const int* in_sizes, int n_in,
                              void* d_out, int out_size, void* d_ws, size_t ws_size,
                              hipStream_t stream) {
    const float* x    = (const float*)d_in[0];
    const float* sw   = (const float*)d_in[1];
    const float* bias = (const float*)d_in[2];
    const int*   rows = (const int*)d_in[3];
    const int*   cols = (const int*)d_in[4];
    const int nnz = in_sizes[1];

    unsigned short* Wswz = (unsigned short*)d_ws;          // 32 MB bf16, frag-linear
    (void)hipMemsetAsync(d_ws, 0, (size_t)CH * CH * sizeof(unsigned short), stream);

    sl_scatter<<<(nnz + 255) / 256, 256, 0, stream>>>(sw, rows, cols, Wswz, nnz);

    // blockIdx.x: N-panel fastest -> each 2MB x M-band reused from L2 16x.
    sl_gemm<<<dim3(NPANELS * MTILES), 256, 0, stream>>>(x, Wswz, bias, (float*)d_out);
}
